// BookRecommendationModel_57492432224464
// MI455X (gfx1250) — matrix operand order repaired and checked by reading, not compiled, not run on hardware
//
#include <hip/hip_runtime.h>
#include <hip/hip_bf16.h>

// ---------------- types / helpers ----------------
typedef __attribute__((ext_vector_type(16))) __bf16 v16bf;
typedef __attribute__((ext_vector_type(8)))  __bf16 v8bf;
typedef __attribute__((ext_vector_type(8)))  float  v8f;
typedef unsigned int u32x4 __attribute__((ext_vector_type(4)));
typedef int          i32x4 __attribute__((ext_vector_type(4)));
typedef int          i32x8 __attribute__((ext_vector_type(8)));

static constexpr int kH   = 768;
static constexpr int kNH  = 8;
static constexpr int kHD  = 96;
static constexpr int kN   = 64;     // "L" for attention
static constexpr int kS   = 512;    // "B" for attention
static constexpr int kTOK = kN * kS;        // 32768
static constexpr int k3H  = 3 * kH;         // 2304
static constexpr int kD   = kS * kH;        // 393216 (kmeans feature dim)
static constexpr float kLNEPS = 1e-5f;

#if defined(__has_builtin)
#if __has_builtin(__builtin_amdgcn_tensor_load_to_lds) && __has_builtin(__builtin_amdgcn_s_wait_tensorcnt)
#define USE_TDM 1
#endif
#endif

__device__ inline v8f zero8() {
  v8f z = {0.f,0.f,0.f,0.f,0.f,0.f,0.f,0.f};
  return z;
}
__device__ inline v8bf ld8(const __bf16* p) { return *(const v8bf*)p; }
__device__ inline v16bf cat16(v8bf lo, v8bf hi) {
  return __builtin_shufflevector(lo, hi, 0,1,2,3,4,5,6,7,8,9,10,11,12,13,14,15);
}
__device__ inline v8f wmma_bf16(v16bf a, v16bf b, v8f c) {
  return __builtin_amdgcn_wmma_f32_16x16x32_bf16(false, a, false, b, (short)0, c, false, false);
}
__device__ inline __bf16 to_bf16(float f) {
  unsigned u = __float_as_uint(f);
  u += 0x7FFFu + ((u >> 16) & 1u);           // round-to-nearest-even
  unsigned short s = (unsigned short)(u >> 16);
  __bf16 r;
  __builtin_memcpy(&r, &s, 2);
  return r;
}

#ifdef USE_TDM
// LDS byte offset of a __shared__ object (generic -> AS3 -> int)
__device__ inline unsigned lds_off_of(const void* p) {
  return (unsigned)(unsigned long long)(__attribute__((address_space(3))) const char*)p;
}
// Issue a TDM 2D tile load: tile_h rows x tile_w elements (2-byte), row stride in elements.
__device__ inline void tdm_load_2d(unsigned lds_off, const void* gptr,
                                   unsigned tile_w, unsigned tile_h,
                                   unsigned long long row_stride_elems) {
  unsigned long long ga = (unsigned long long)gptr;
  u32x4 g0;
  g0.x = 1u;                                             // count=1, user mode
  g0.y = lds_off;                                        // lds_addr
  g0.z = (unsigned)(ga & 0xFFFFFFFFu);                   // global_addr[31:0]
  g0.w = (unsigned)((ga >> 32) & 0x01FFFFFFu) | 0x80000000u; // addr[56:32] | type=2
  i32x8 g1;
  g1[0] = (int)(1u << 16);                               // data_size = 2 bytes
  g1[1] = (int)(tile_w << 16);                           // tensor_dim0[15:0]
  g1[2] = (int)(tile_h << 16);                           // dim0 hi=0 | tensor_dim1[15:0]
  g1[3] = (int)(tile_w << 16);                           // dim1 hi=0 | tile_dim0
  g1[4] = (int)(tile_h & 0xFFFF);                        // tile_dim1 | tile_dim2=0
  g1[5] = (int)(row_stride_elems & 0xFFFFFFFFu);         // dim0_stride[31:0]
  g1[6] = (int)((row_stride_elems >> 32) & 0xFFFF);      // dim0_stride[47:32] | dim1_stride lo=0
  g1[7] = 0;
  i32x4 z4 = {0, 0, 0, 0};
#if __clang_major__ >= 23
  i32x8 z8 = {0, 0, 0, 0, 0, 0, 0, 0};
  __builtin_amdgcn_tensor_load_to_lds(g0, g1, z4, z4, z8, 0);
#else
  __builtin_amdgcn_tensor_load_to_lds(g0, g1, z4, z4, 0);
#endif
}
#endif

// ---------------- f32 -> bf16 convert ----------------
__global__ __launch_bounds__(256) void k_cvt_bf16(const float* __restrict__ s,
                                                  __bf16* __restrict__ d, int n) {
  int i = blockIdx.x * 256 + threadIdx.x;
  if (i < n) d[i] = to_bf16(s[i]);
}

// ---------------- embedding gather + LayerNorm ----------------
__global__ __launch_bounds__(256) void k_embed_ln(const int* __restrict__ ids,
    const float* __restrict__ wemb, const float* __restrict__ pemb,
    const float* __restrict__ gamma, const float* __restrict__ beta,
    __bf16* __restrict__ x) {
  __shared__ float red[256];
  __shared__ float s_mu, s_rstd;
  int tok = blockIdx.x;
  int s   = tok & (kS - 1);
  int tid = threadIdx.x;
  int id  = ids[tok];
  const float* wr = wemb + (size_t)id * kH;
  const float* pr = pemb + (size_t)s  * kH;
  float v0 = wr[tid]       + pr[tid];
  float v1 = wr[tid + 256] + pr[tid + 256];
  float v2 = wr[tid + 512] + pr[tid + 512];
  red[tid] = v0 + v1 + v2;
  __syncthreads();
  for (int off = 128; off > 0; off >>= 1) { if (tid < off) red[tid] += red[tid + off]; __syncthreads(); }
  if (tid == 0) s_mu = red[0] * (1.f / (float)kH);
  __syncthreads();
  float mu = s_mu;
  float d0 = v0 - mu, d1 = v1 - mu, d2 = v2 - mu;
  red[tid] = d0*d0 + d1*d1 + d2*d2;
  __syncthreads();
  for (int off = 128; off > 0; off >>= 1) { if (tid < off) red[tid] += red[tid + off]; __syncthreads(); }
  if (tid == 0) s_rstd = rsqrtf(red[0] * (1.f / (float)kH) + kLNEPS);
  __syncthreads();
  float rs = s_rstd;
  __bf16* xr = x + (size_t)tok * kH;
  xr[tid]       = to_bf16(d0 * rs * gamma[tid]       + beta[tid]);
  xr[tid + 256] = to_bf16(d1 * rs * gamma[tid + 256] + beta[tid + 256]);
  xr[tid + 512] = to_bf16(d2 * rs * gamma[tid + 512] + beta[tid + 512]);
}

// ---------------- WMMA GEMM: C[M,N] = A[M,K] * W[N,K]^T + bias ----------------
// 256 threads = 8 waves (2 M x 4 N); wave tile 32x64; block tile 64x256.
// Software-pipelined: double-buffered fragment registers.
struct Frag { v16bf a0, a1, b0, b1, b2, b3; };

template <int K, bool OUTF32>
__global__ __launch_bounds__(256) void k_gemm(const __bf16* __restrict__ A,
    const __bf16* __restrict__ Bw, const float* __restrict__ bias,
    void* __restrict__ Out, int N) {
  int tid  = threadIdx.x;
  int wave = tid >> 5, lane = tid & 31;
  int wm = wave & 1, wn = wave >> 1;
  int rowBase = blockIdx.x * 64  + wm * 32;
  int colBase = blockIdx.y * 256 + wn * 64;
  int mrow = lane & 15;
  int hi16 = (lane >> 4) & 1;
  int klo  = hi16 * 16;    // A K offset per lane group
  int kbO  = hi16 * 16;    // B K offset per lane group

  const __bf16* ar0 = A + (size_t)(rowBase + mrow) * K + klo;
  const __bf16* ar1 = ar0 + (size_t)16 * K;
  const __bf16* bp0 = Bw + (size_t)(colBase + mrow) * K + kbO;
  const __bf16* bp1 = bp0 + (size_t)16 * K;
  const __bf16* bp2 = bp0 + (size_t)32 * K;
  const __bf16* bp3 = bp0 + (size_t)48 * K;

  v8f acc[8];
  #pragma unroll
  for (int t = 0; t < 8; ++t) acc[t] = zero8();

  auto loadf = [&](Frag& f, int k) {
    f.a0 = cat16(ld8(ar0 + k), ld8(ar0 + k + 8));
    f.a1 = cat16(ld8(ar1 + k), ld8(ar1 + k + 8));
    f.b0 = cat16(ld8(bp0 + k), ld8(bp0 + k + 8));
    f.b1 = cat16(ld8(bp1 + k), ld8(bp1 + k + 8));
    f.b2 = cat16(ld8(bp2 + k), ld8(bp2 + k + 8));
    f.b3 = cat16(ld8(bp3 + k), ld8(bp3 + k + 8));
  };
  auto mma = [&](const Frag& f) {
    acc[0] = wmma_bf16(f.a0, f.b0, acc[0]);
    acc[1] = wmma_bf16(f.a0, f.b1, acc[1]);
    acc[2] = wmma_bf16(f.a0, f.b2, acc[2]);
    acc[3] = wmma_bf16(f.a0, f.b3, acc[3]);
    acc[4] = wmma_bf16(f.a1, f.b0, acc[4]);
    acc[5] = wmma_bf16(f.a1, f.b1, acc[5]);
    acc[6] = wmma_bf16(f.a1, f.b2, acc[6]);
    acc[7] = wmma_bf16(f.a1, f.b3, acc[7]);
  };

  Frag f0, f1;
  loadf(f0, 0);
  static_assert(K % 64 == 0, "K must be multiple of 64");
  for (int k = 0; k < K - 64; k += 64) {
    loadf(f1, k + 32);
    mma(f0);
    loadf(f0, k + 64);
    mma(f1);
  }
  loadf(f1, K - 32);
  mma(f0);
  mma(f1);

  #pragma unroll
  for (int half = 0; half < 2; ++half) {
    #pragma unroll
    for (int t = 0; t < 4; ++t) {
      int col = colBase + t * 16 + mrow;
      float bv = bias[col];
      #pragma unroll
      for (int r = 0; r < 8; ++r) {
        int row = rowBase + half * 16 + r + hi16 * 8;
        float v = acc[half * 4 + t][r] + bv;
        if (OUTF32) ((float*)Out)[(size_t)row * N + col] = v;
        else        ((__bf16*)Out)[(size_t)row * N + col] = to_bf16(v);
      }
    }
  }
}

// ---------------- attention: one block per (b, h); L=64, HD=96 ----------------
__global__ __launch_bounds__(256) void k_attn(const __bf16* __restrict__ qkv,
                                              __bf16* __restrict__ o) {
  __shared__ __align__(16) char smem[61440];
  __bf16* sQ  = (__bf16*)(smem);            // 64x96 bf16  (12288 B)
  __bf16* sK  = (__bf16*)(smem + 12288);    // 64x96 bf16  (12288 B)
  __bf16* sVt = (__bf16*)(smem + 24576);    // 96x64 bf16  (12288 B)  [d][m]
  float*  sS  = (float*) (smem + 36864);    // 64x64 f32   (16384 B)
  __bf16* sV  = (__bf16*)(smem + 36864);    // alias: staged V (TDM path only)
  __bf16* sA  = (__bf16*)(smem + 53248);    // 64x64 bf16  (8192 B)

  int b = blockIdx.x & (kS - 1);
  int h = blockIdx.x >> 9;
  int tid = threadIdx.x;

#ifdef USE_TDM
  if (tid < 32) {  // wave 0 issues the DMAs (TDM ignores EXEC; 1 issue per wave)
    unsigned base = lds_off_of(smem);
    const __bf16* g = qkv + (size_t)b * k3H + h * kHD;   // row l=0
    unsigned long long stride = (unsigned long long)kS * k3H;  // elems between rows
    tdm_load_2d(base + 0,     g,          96, 64, stride);   // Q
    tdm_load_2d(base + 12288, g + kH,     96, 64, stride);   // K
    tdm_load_2d(base + 36864, g + 2*kH,   96, 64, stride);   // V (row-major stage)
    __builtin_amdgcn_s_wait_tensorcnt(0);
  }
  __syncthreads();
  // transpose V: sV[l][d] -> sVt[d][l]
  for (int i = tid; i < 64 * 96; i += 256) {
    int l = i / 96, d = i % 96;
    sVt[d * 64 + l] = sV[l * 96 + d];
  }
  __syncthreads();
#else
  for (int i = tid; i < 768; i += 256) {
    int l = i / 12, c = i % 12;
    const __bf16* base = qkv + (size_t)(l * kS + b) * k3H + h * kHD;
    *((uint4*)(sQ + l * 96) + c) = *((const uint4*)base + c);
    *((uint4*)(sK + l * 96) + c) = *((const uint4*)(base + kH) + c);
  }
  for (int i = tid; i < 64 * 96; i += 256) {
    int l = i / 96, d = i % 96;
    sVt[d * 64 + l] = qkv[(size_t)(l * kS + b) * k3H + 2 * kH + h * kHD + d];
  }
  __syncthreads();
#endif

  int wave = tid >> 5, lane = tid & 31;
  int nlo  = lane & 15;
  int hi16 = (lane >> 4) & 1;
  int klo  = hi16 * 16, kbO = hi16 * 16;
  const float scale = 0.10206207261596577f;   // 1/sqrt(96)

  // scores = Q * K^T : 16 tiles of 16x16; each wave does 2 (same row, A reuse)
  {
    int mi  = wave >> 1;
    int ni0 = (wave & 1) * 2;
    int m   = mi * 16 + nlo;
    v8f a0 = zero8(), a1 = zero8();
    #pragma unroll
    for (int kk = 0; kk < 96; kk += 32) {
      v16bf af = cat16(ld8(sQ + m * 96 + kk + klo), ld8(sQ + m * 96 + kk + klo + 8));
      const __bf16* q0 = sK + (ni0 * 16 + nlo) * 96 + kk + kbO;
      const __bf16* q1 = sK + ((ni0 + 1) * 16 + nlo) * 96 + kk + kbO;
      a0 = wmma_bf16(af, cat16(ld8(q0), ld8(q0 + 8)), a0);
      a1 = wmma_bf16(af, cat16(ld8(q1), ld8(q1 + 8)), a1);
    }
    #pragma unroll
    for (int r = 0; r < 8; ++r) {
      int row = mi * 16 + r + hi16 * 8;
      sS[row * 64 + ni0 * 16 + nlo]       = a0[r] * scale;
      sS[row * 64 + (ni0 + 1) * 16 + nlo] = a1[r] * scale;
    }
  }
  __syncthreads();

  // softmax over rows (64 rows, one per thread 0..63)
  if (tid < 64) {
    float* row = sS + tid * 64;
    float mx = row[0];
    for (int j = 1; j < 64; ++j) mx = fmaxf(mx, row[j]);
    float sum = 0.f;
    for (int j = 0; j < 64; ++j) { float e = __expf(row[j] - mx); row[j] = e; sum += e; }
    float inv = 1.f / sum;
    for (int j = 0; j < 64; ++j) sA[tid * 64 + j] = to_bf16(row[j] * inv);
  }
  __syncthreads();

  // O = attn(64x64) * V(64x96): 24 tiles; each wave does 3
  for (int tt = wave * 3; tt < wave * 3 + 3; ++tt) {
    int mi = tt / 6, ni = tt % 6;
    int m  = mi * 16 + nlo;
    v8f acc = zero8();
    #pragma unroll
    for (int kk = 0; kk < 64; kk += 32) {
      v16bf af = cat16(ld8(sA + m * 64 + kk + klo), ld8(sA + m * 64 + kk + klo + 8));
      const __bf16* bp = sVt + (ni * 16 + nlo) * 64 + kk + kbO;
      acc = wmma_bf16(af, cat16(ld8(bp), ld8(bp + 8)), acc);
    }
    #pragma unroll
    for (int r = 0; r < 8; ++r) {
      int l = mi * 16 + r + hi16 * 8;
      o[(size_t)(l * kS + b) * kH + h * kHD + ni * 16 + nlo] = to_bf16(acc[r]);
    }
  }
}

// ---------------- kmeans: centroid norms (float4 vectorized) ----------------
__global__ __launch_bounds__(256) void k_c2(const float* __restrict__ cent,
                                            float* __restrict__ c2) {
  __shared__ float red[256];
  int k = blockIdx.x, tid = threadIdx.x;
  const float* cr = cent + (size_t)k * kD;
  float acc = 0.f;
  for (int d = tid * 4; d < kD; d += 1024) {
    float4 c = *(const float4*)(cr + d);
    acc += c.x*c.x + c.y*c.y + c.z*c.z + c.w*c.w;
  }
  red[tid] = acc; __syncthreads();
  for (int off = 128; off > 0; off >>= 1) { if (tid < off) red[tid] += red[tid + off]; __syncthreads(); }
  if (tid == 0) c2[k] = red[0];
}

// ---------------- kmeans: dots + argmin (float4 vectorized) ----------------
__global__ __launch_bounds__(256) void k_assign(const float* __restrict__ flat,
    const float* __restrict__ cent, const float* __restrict__ c2,
    float* __restrict__ cl_out, int* __restrict__ cl_int) {
  __shared__ float red[256];
  __shared__ float sv[11];
  int n = blockIdx.x, tid = threadIdx.x;
  const float* fr = flat + (size_t)n * kD;
  float dot[10];
  #pragma unroll
  for (int k = 0; k < 10; ++k) dot[k] = 0.f;
  float x2 = 0.f;
  for (int d = tid * 4; d < kD; d += 1024) {
    __builtin_prefetch(fr + d + 4096, 0, 0);
    float4 f = *(const float4*)(fr + d);
    x2 += f.x*f.x + f.y*f.y + f.z*f.z + f.w*f.w;
    #pragma unroll
    for (int k = 0; k < 10; ++k) {
      float4 c = *(const float4*)(cent + (size_t)k * kD + d);
      dot[k] += f.x*c.x + f.y*c.y + f.z*c.z + f.w*c.w;
    }
  }
  for (int k = 0; k < 11; ++k) {
    float v = (k < 10) ? dot[k] : x2;
    red[tid] = v; __syncthreads();
    for (int off = 128; off > 0; off >>= 1) { if (tid < off) red[tid] += red[tid + off]; __syncthreads(); }
    if (tid == 0) sv[k] = red[0];
    __syncthreads();
  }
  if (tid == 0) {
    float best = 3.4e38f; int bi = 0;
    for (int k = 0; k < 10; ++k) {
      float dd = sv[10] + c2[k] - 2.f * sv[k];
      if (dd < best) { best = dd; bi = k; }
    }
    cl_out[n] = (float)bi;
    cl_int[n] = bi;
  }
}

// ---------------- kmeans: per-row loss partial (float4 vectorized) ----------------
__global__ __launch_bounds__(256) void k_loss(const float* __restrict__ flat,
    const float* __restrict__ cent, const int* __restrict__ cl,
    float* __restrict__ partial) {
  __shared__ float red[256];
  int n = blockIdx.x, tid = threadIdx.x;
  const float* fr = flat + (size_t)n * kD;
  const float* cr = cent + (size_t)cl[n] * kD;
  float acc = 0.f;
  for (int d = tid * 4; d < kD; d += 1024) {
    float4 f = *(const float4*)(fr + d);
    float4 c = *(const float4*)(cr + d);
    float dx = f.x - c.x, dy = f.y - c.y, dz = f.z - c.z, dw = f.w - c.w;
    acc += dx*dx + dy*dy + dz*dz + dw*dw;
  }
  red[tid] = acc; __syncthreads();
  for (int off = 128; off > 0; off >>= 1) { if (tid < off) red[tid] += red[tid + off]; __syncthreads(); }
  if (tid == 0) partial[n] = red[0];
}

__global__ __launch_bounds__(64) void k_loss_final(const float* __restrict__ partial,
                                                   float* __restrict__ loss) {
  __shared__ float red[64];
  int tid = threadIdx.x;
  red[tid] = partial[tid]; __syncthreads();
  for (int off = 32; off > 0; off >>= 1) { if (tid < off) red[tid] += red[tid + off]; __syncthreads(); }
  if (tid == 0) *loss = red[0];
}

// ---------------- launch ----------------
extern "C" void kernel_launch(void* const* d_in, const int* in_sizes, int n_in,
                              void* d_out, int out_size, void* d_ws, size_t ws_size,
                              hipStream_t stream) {
  const int*   ids  = (const int*)  d_in[0];
  const float* wemb = (const float*)d_in[1];
  const float* pemb = (const float*)d_in[2];
  const float* gam  = (const float*)d_in[3];
  const float* bet  = (const float*)d_in[4];
  const float* win  = (const float*)d_in[5];
  const float* bin  = (const float*)d_in[6];
  const float* wout = (const float*)d_in[7];
  const float* bout = (const float*)d_in[8];
  const float* cent = (const float*)d_in[9];

  char* ws = (char*)d_ws;
  __bf16* x_bf    = (__bf16*)(ws + 0);            // 32768*768*2   = 50331648
  __bf16* win_bf  = (__bf16*)(ws + 50331648);     // 2304*768*2    =  3538944
  __bf16* wout_bf = (__bf16*)(ws + 53870592);     // 768*768*2     =  1179648
  __bf16* qkv_bf  = (__bf16*)(ws + 55050240);     // 32768*2304*2  = 150994944
  __bf16* o_bf    = (__bf16*)(ws + 206045184);    // 32768*768*2   = 50331648
  float*  attn    = (float*) (ws + 256376832);    // 32768*768*4   = 100663296
  float*  c2      = (float*) (ws + 357040128);    // 10*4 (padded)
  int*    cli     = (int*)   (ws + 357040192);    // 64*4 (padded)
  float*  part    = (float*) (ws + 357040448);    // 64*4

  float* outf = (float*)d_out;

  k_cvt_bf16<<<(k3H * kH + 255) / 256, 256, 0, stream>>>(win, win_bf, k3H * kH);
  k_cvt_bf16<<<(kH * kH + 255) / 256, 256, 0, stream>>>(wout, wout_bf, kH * kH);
  k_embed_ln<<<kTOK, 256, 0, stream>>>(ids, wemb, pemb, gam, bet, x_bf);
  k_gemm<768, false><<<dim3(kTOK / 64, k3H / 256), 256, 0, stream>>>(x_bf, win_bf, bin,
                                                                     qkv_bf, k3H);
  k_attn<<<kS * kNH, 256, 0, stream>>>(qkv_bf, o_bf);
  k_gemm<768, true><<<dim3(kTOK / 64, kH / 256), 256, 0, stream>>>(o_bf, wout_bf, bout,
                                                                   attn, kH);
  k_c2<<<10, 256, 0, stream>>>(cent, c2);
  k_assign<<<kN, 256, 0, stream>>>(attn, cent, c2, outf, cli);
  k_loss<<<kN, 256, 0, stream>>>(attn, cent, cli, part);
  k_loss_final<<<1, 64, 0, stream>>>(part, outf + 64);
}
